// AttentionBlockManual_46634754900675
// MI455X (gfx1250) — compile-verified, matrix operand order checked by reading, not run on hardware
//
#include <hip/hip_runtime.h>
#include <hip/hip_bf16.h>
#include <math.h>

typedef _Float16 half8  __attribute__((ext_vector_type(8)));
typedef _Float16 half16 __attribute__((ext_vector_type(16)));
typedef float    float8 __attribute__((ext_vector_type(8)));

#define N_B 32
#define C_DIM 256
#define HW 32
#define S_TOK 1024   // 32*32 tokens per batch
#define D_DIM 256
#define T_DIM 512
#define SCALE 0.0625f  // 1/sqrt(256)

#define WSZ (256*256)          // one weight matrix, elements
#define QSZ (N_B*S_TOK*D_DIM)  // one activation tensor, elements

// Load a 16x32 f16 A/B fragment from a row-major matrix.
// p must already point at  base + row*ld + ((lane>>4)*8); k0 is the K offset.
__device__ __forceinline__ half16 ldfrag(const _Float16* p, int k0) {
    union { half16 v; half8 h[2]; } u;
    u.h[0] = *(const half8*)(p + k0);
    u.h[1] = *(const half8*)(p + k0 + 16);
    return u.v;
}

__device__ __forceinline__ float8 wmma_f16(half16 a, half16 b, float8 c) {
    return __builtin_amdgcn_wmma_f32_16x16x32_f16(false, a, false, b, (short)0, c, false, false);
}

// ---------------- kernel 1: convert weights f32 -> f16 ----------------
__global__ void cvt_weights_kernel(const float* Wq, const float* Wk,
                                   const float* Wv, const float* Wp,
                                   _Float16* dst) {
    int idx = blockIdx.x * blockDim.x + threadIdx.x;
    if (idx >= 4 * WSZ) return;
    int which = idx >> 16;          // 65536 per matrix
    int w = idx & (WSZ - 1);
    const float* src = (which == 0) ? Wq : (which == 1) ? Wk : (which == 2) ? Wv : Wp;
    dst[idx] = (_Float16)src[w];
}

// ---------------- kernel 2: temb = relu(t @ W_t^T + b_t) ----------------
__global__ void temb_kernel(const float* t, const float* W_t, const float* b_t,
                            float* temb) {
    int n = blockIdx.x;          // 0..31
    int c = threadIdx.x;         // 0..255
    const float* tr = t + n * T_DIM;
    const float* wr = W_t + c * T_DIM;
    float acc = b_t[c];
    #pragma unroll 8
    for (int j = 0; j < T_DIM; ++j) acc += tr[j] * wr[j];
    temb[n * C_DIM + c] = acc > 0.f ? acc : 0.f;
}

// ---------------- kernel 3: xc = x + temb ; q,k,vT projections ----------------
// grid: 32*64 blocks, 128 threads (4 waves). Each block = one 16-token tile.
__global__ void qkv_kernel(const float* x, const float* temb,
                           const _Float16* Wq16, const _Float16* Wk16,
                           const _Float16* Wv16,
                           _Float16* q16, _Float16* k16, _Float16* vT16) {
    __shared__ __align__(16) _Float16 xtile[16][C_DIM];   // 8 KB

    const int n = blockIdx.x >> 6;
    const int sbase = (blockIdx.x & 63) * 16;
    const int tid = threadIdx.x;

    // build (x + temb) tile, transposed to [token][channel], f16
    for (int e = tid; e < 16 * C_DIM; e += 128) {
        int tok = e >> 8, c = e & 255;
        int s = sbase + tok, h = s & 31, w = s >> 5;
        float v = x[((n * C_DIM + c) * HW + h) * HW + w] + temb[n * C_DIM + c];
        xtile[tok][c] = (_Float16)v;
    }
    __syncthreads();

    const int lane = tid & 31;
    const int wv = tid >> 5;              // 0..3
    const int hb8 = (lane >> 4) * 8;
    const int ln15 = lane & 15;

    // xtile fragments: act as A (rows=tokens) for q/k, and as B (cols=tokens) for vT
    half16 xa[8];
    {
        const _Float16* xp = &xtile[ln15][hb8];
        #pragma unroll
        for (int cs = 0; cs < 8; ++cs) xa[cs] = ldfrag(xp, cs * 32);
    }

    // 48 output tiles: 16 for q, 16 for k, 16 for vT
    for (int idx = wv; idx < 48; idx += 4) {
        const int mat = idx >> 4;         // 0=q 1=k 2=v
        const int dt = idx & 15;
        float8 acc = {};
        if (mat < 2) {
            const _Float16* Wm = (mat == 0) ? Wq16 : Wk16;
            const _Float16* bp = Wm + (dt * 16 + ln15) * C_DIM + hb8; // B: W[d][c], col=d
            #pragma unroll
            for (int cs = 0; cs < 8; ++cs)
                acc = wmma_f16(xa[cs], ldfrag(bp, cs * 32), acc);
            _Float16* out = (mat == 0) ? q16 : k16;
            #pragma unroll
            for (int r = 0; r < 8; ++r) {
                int tok = r + hb8;
                out[(((size_t)n * S_TOK + sbase + tok) * D_DIM) + dt * 16 + ln15] =
                    (_Float16)acc[r];
            }
        } else {
            // vT = Wv * xc^T : A = Wv rows (d), B = xc^T (col = token)
            const _Float16* ap = Wv16 + (dt * 16 + ln15) * C_DIM + hb8;
            #pragma unroll
            for (int cs = 0; cs < 8; ++cs)
                acc = wmma_f16(ldfrag(ap, cs * 32), xa[cs], acc);
            #pragma unroll
            for (int r = 0; r < 8; ++r) {
                int dd = dt * 16 + r + hb8;
                vT16[(((size_t)n * D_DIM + dd) * S_TOK) + sbase + ln15] = (_Float16)acc[r];
            }
        }
    }
}

// ---------------- kernel 4: fused attention for one (n, 16-query tile) ----------------
// grid: 32*64 blocks, 256 threads (8 waves). LDS ~33.3 KB.
__global__ void attn_kernel(const _Float16* q16, const _Float16* k16,
                            const _Float16* vT16, _Float16* o16) {
    __shared__ __align__(16) _Float16 p16[16][S_TOK];   // 32 KB exp-probs (f16)
    __shared__ float psum[256];
    __shared__ float rowsum[16];

    const int n = blockIdx.x >> 6;
    const int sqbase = (blockIdx.x & 63) * 16;
    const int tid = threadIdx.x;
    const int lane = tid & 31;
    const int wv = tid >> 5;              // 0..7
    const int hb8 = (lane >> 4) * 8;
    const int ln15 = lane & 15;

    // preload the 8 Q fragments (shared by all 8 key tiles of this wave)
    half16 qa[8];
    {
        const _Float16* ap = q16 + (((size_t)n * S_TOK + sqbase + ln15) * D_DIM) + hb8;
        #pragma unroll
        for (int cs = 0; cs < 8; ++cs) qa[cs] = ldfrag(ap, cs * 32);
    }

    // ---- QK^T : wave wv owns keys [wv*128, wv*128+128); exp applied at store ----
    for (int kt = 0; kt < 8; ++kt) {
        const int keybase = wv * 128 + kt * 16;
        const _Float16* bp = k16 + (((size_t)n * S_TOK + keybase + ln15) * D_DIM) + hb8;
        float8 acc = {};
        #pragma unroll
        for (int cs = 0; cs < 8; ++cs)
            acc = wmma_f16(qa[cs], ldfrag(bp, cs * 32), acc);
        #pragma unroll
        for (int r = 0; r < 8; ++r)
            p16[r + hb8][keybase + ln15] = (_Float16)__expf(acc[r] * SCALE);
    }
    __syncthreads();

    // ---- row sums over f16 probs ----
    {
        const int row = tid >> 4;
        const int cb = (tid & 15) * 64;
        float s = 0.f;
        #pragma unroll
        for (int j = 0; j < 8; ++j) {
            half8 h = *(const half8*)&p16[row][cb + j * 8];
            #pragma unroll
            for (int i = 0; i < 8; ++i) s += (float)h[i];
        }
        psum[tid] = s;
    }
    __syncthreads();
    if (tid < 16) {
        float s = 0.f;
        #pragma unroll
        for (int j = 0; j < 16; ++j) s += psum[tid * 16 + j];
        rowsum[tid] = s;
    }
    __syncthreads();

    // ---- P @ V : one A-fragment feeds both of this wave's d-tiles ----
    {
        const _Float16* pa = &p16[ln15][hb8];
        const _Float16* b0 = vT16 + (((size_t)n * D_DIM + wv * 16 + ln15) * S_TOK) + hb8;
        const _Float16* b1 = vT16 + (((size_t)n * D_DIM + (wv + 8) * 16 + ln15) * S_TOK) + hb8;
        float8 acc0 = {}, acc1 = {};
        for (int kb = 0; kb < S_TOK; kb += 32) {
            half16 a = ldfrag(pa, kb);
            acc0 = wmma_f16(a, ldfrag(b0, kb), acc0);
            acc1 = wmma_f16(a, ldfrag(b1, kb), acc1);
        }
        #pragma unroll
        for (int r = 0; r < 8; ++r) {
            int qrow = r + hb8;
            float inv = 1.f / rowsum[qrow];
            size_t base = ((size_t)n * S_TOK + sqbase + qrow) * D_DIM + ln15;
            o16[base + wv * 16]       = (_Float16)(acc0[r] * inv);
            o16[base + (wv + 8) * 16] = (_Float16)(acc1[r] * inv);
        }
    }
}

// ---------------- kernel 5: out = transpose(o @ Wp^T + bp) + x ----------------
// grid: 32*64 blocks, 128 threads (4 waves).
__global__ void proj_kernel(const _Float16* o16, const _Float16* Wp16,
                            const float* bp, const float* x, float* out) {
    const int n = blockIdx.x >> 6;
    const int sbase = (blockIdx.x & 63) * 16;
    const int tid = threadIdx.x;
    const int lane = tid & 31;
    const int wv = tid >> 5;              // 0..3
    const int hb8 = (lane >> 4) * 8;
    const int ln15 = lane & 15;

    // preload o-tile A fragments (shared across the 4 c-tiles of this wave)
    half16 oa[8];
    {
        const _Float16* ap = o16 + (((size_t)n * S_TOK + sbase + ln15) * D_DIM) + hb8;
        #pragma unroll
        for (int ds = 0; ds < 8; ++ds) oa[ds] = ldfrag(ap, ds * 32);
    }

    for (int ct = wv; ct < 16; ct += 4) {
        const int ccol = ct * 16 + ln15;
        const _Float16* bw = Wp16 + ccol * D_DIM + hb8;
        float8 acc = {};
        #pragma unroll
        for (int ds = 0; ds < 8; ++ds)
            acc = wmma_f16(oa[ds], ldfrag(bw, ds * 32), acc);
        #pragma unroll
        for (int r = 0; r < 8; ++r) {
            int s = sbase + r + hb8;
            int h = s & 31, w = s >> 5;
            size_t addr = (((size_t)n * C_DIM + ccol) * HW + h) * HW + w;
            out[addr] = acc[r] + bp[ccol] + x[addr];
        }
    }
}

extern "C" void kernel_launch(void* const* d_in, const int* in_sizes, int n_in,
                              void* d_out, int out_size, void* d_ws, size_t ws_size,
                              hipStream_t stream) {
    const float* x   = (const float*)d_in[0];
    const float* t   = (const float*)d_in[1];
    const float* W_t = (const float*)d_in[2];
    const float* b_t = (const float*)d_in[3];
    const float* Wq  = (const float*)d_in[4];
    const float* Wk  = (const float*)d_in[5];
    const float* Wv  = (const float*)d_in[6];
    const float* Wp  = (const float*)d_in[7];
    const float* bp  = (const float*)d_in[8];
    float* out = (float*)d_out;

    // workspace layout (all f16 regions first, then f32 temb)
    _Float16* w16  = (_Float16*)d_ws;
    _Float16* Wq16 = w16;
    _Float16* Wk16 = w16 + WSZ;
    _Float16* Wv16 = w16 + 2 * WSZ;
    _Float16* Wp16 = w16 + 3 * WSZ;
    _Float16* q16  = w16 + 4 * WSZ;
    _Float16* k16  = q16 + QSZ;
    _Float16* vT16 = k16 + QSZ;
    _Float16* o16  = vT16 + QSZ;
    float* temb = (float*)(o16 + QSZ);

    cvt_weights_kernel<<<(4 * WSZ + 255) / 256, 256, 0, stream>>>(Wq, Wk, Wv, Wp, w16);
    temb_kernel<<<N_B, C_DIM, 0, stream>>>(t, W_t, b_t, temb);
    qkv_kernel<<<N_B * 64, 128, 0, stream>>>(x, temb, Wq16, Wk16, Wv16, q16, k16, vT16);
    attn_kernel<<<N_B * 64, 256, 0, stream>>>(q16, k16, vT16, o16);
    proj_kernel<<<N_B * 64, 128, 0, stream>>>(o16, Wp16, bp, x, out);
}
